// CNUs_39642548142128
// MI455X (gfx1250) — matrix operand order repaired and checked by reading, not compiled, not run on hardware
//
#include <hip/hip_runtime.h>

typedef __attribute__((ext_vector_type(16))) __bf16 v16bf;
typedef __attribute__((ext_vector_type(8)))  __bf16 v8bf;
typedef __attribute__((ext_vector_type(8)))  float  v8f;

#define QN     32
#define DN     64
#define MKEYS  4096
#define UN     64
#define DELTA  16
#define BN     1024
#define NTILE  (MKEYS / 16)

#define KEY_VMASK 0xFFFFF000u
#define KEY_IMASK 0x00000FFFu

static __device__ __forceinline__ unsigned short f2bf(float f) {
    unsigned u = __float_as_uint(f);
    unsigned r = (u + 0x7FFFu + ((u >> 16) & 1u)) >> 16;   // round-to-nearest-even
    return (unsigned short)r;
}
static __device__ __forceinline__ unsigned umaxu(unsigned a, unsigned b) { return a > b ? a : b; }
static __device__ __forceinline__ unsigned uminu(unsigned a, unsigned b) { return a < b ? a : b; }

union Frag {                       // two b128 loads alias the WMMA operand
    v16bf v;
    v8bf  h[2];
};

// A-operand fragment layout (ISA 16-bit A 16x32): per lane, two 16B chunks
// at half-offsets {khi, khi+16} where khi = (lane<16 ? 0 : 8).
static __device__ __forceinline__ void loadA(const unsigned short* p, Frag& f0, Frag& f1) {
    f0.h[0] = *(const v8bf*)(p);        // K khi..khi+7
    f0.h[1] = *(const v8bf*)(p + 16);   // K khi+16..khi+23
    f1.h[0] = *(const v8bf*)(p + 32);   // second 16x32 fragment (K+32)
    f1.h[1] = *(const v8bf*)(p + 48);
}

// ---------------------------------------------------------------------------
// Kernel 1: L2-normalize rows of x [B,D] and K [Q*M,D], write bf16 to ws.
// ---------------------------------------------------------------------------
__global__ __launch_bounds__(256)
void cnu_normalize(const float* __restrict__ x,
                   const float* __restrict__ K,
                   unsigned short* __restrict__ wsx,
                   unsigned short* __restrict__ wsK) {
    const int wave = threadIdx.x >> 5;
    const int lane = threadIdx.x & 31;
    const int row  = blockIdx.x * 8 + wave;
    const int total = BN + QN * MKEYS;
    if (row >= total) return;

    const float* src;
    unsigned short* dst;
    if (row < BN) {
        src = x + (size_t)row * DN;
        dst = wsx + (size_t)row * DN;
    } else {
        const int rr = row - BN;
        src = K + (size_t)rr * DN;
        dst = wsK + (size_t)rr * DN;
    }

    float2 v = ((const float2*)src)[lane];
    float ss = v.x * v.x + v.y * v.y;
#pragma unroll
    for (int off = 16; off > 0; off >>= 1)
        ss += __shfl_xor(ss, off, 32);
    const float inv = 1.0f / fmaxf(sqrtf(ss), 1e-12f);

    const unsigned short h0 = f2bf(v.x * inv);
    const unsigned short h1 = f2bf(v.y * inv);
    ((unsigned int*)dst)[lane] = (unsigned)h0 | ((unsigned)h1 << 16);
}

// One pipeline step: 2x WMMA on this buffer, immediately re-fill the buffer
// with tile+2 (ping-pong double buffering, no rotation copies), then the
// branch-free packed-key top-16 update (v_maxmin_u32 chain).
static __device__ __forceinline__ void stepTile(int tile,
                                                Frag& A0, Frag& A1,
                                                const Frag& B0, const Frag& B1,
                                                const v8f& two,
                                                const unsigned short* aLane,  // aBase + r16*DN (+khi)
                                                int mAdd,
                                                unsigned key[DELTA]) {
    // C seeded with +2.0 bias: WMMA emits resp+2 (positive -> u32-orderable bits)
    v8f acc = __builtin_amdgcn_wmma_f32_16x16x32_bf16(
                  false, A0.v, false, B0.v, (short)0, two, false, false);
    acc = __builtin_amdgcn_wmma_f32_16x16x32_bf16(
                  false, A1.v, false, B1.v, (short)0, acc, false, false);

    // refill this buffer with tile+2 (wrap avoids a branch; extra reload is harmless)
    const int nt = (tile + 2) & (NTILE - 1);
    const unsigned short* aNext = aLane + (size_t)nt * (16 * DN);
    __builtin_prefetch(aNext + 32 * DN, 0, 1);   // tile+4 -> global_prefetch_b8
    loadA(aNext, A0, A1);

    const unsigned mBase = (unsigned)(tile * 16 + mAdd);
#pragma unroll
    for (int r = 0; r < 8; ++r) {
        const unsigned ku = (__float_as_uint(acc[r]) & KEY_VMASK) | (mBase + (unsigned)r);
        if (ku > key[DELTA - 1]) {
            // sorted-desc insertion == med3 clamp chain -> v_maxmin_u32
#pragma unroll
            for (int j = DELTA - 1; j >= 1; --j)
                key[j] = uminu(key[j - 1], umaxu(ku, key[j]));
            key[0] = umaxu(ku, key[0]);
        }
    }
}

// ---------------------------------------------------------------------------
// Kernel 2: one wave per (q, 16-batch tile).
//   A = Kn tile [16 m x 64 d] bf16, B = xn tile [64 d x 16 b] bf16.
// ---------------------------------------------------------------------------
__global__ __launch_bounds__(32)
void cnu_main(const unsigned short* __restrict__ wsx,
              const unsigned short* __restrict__ wsK,
              const float* __restrict__ Mm,
              float* __restrict__ out) {
    const int lane = threadIdx.x;            // 0..31
    const int b0   = blockIdx.x * 16;        // batch tile base
    const int q    = blockIdx.y;
    const int r16  = lane & 15;
    const int khi  = (lane < 16) ? 0 : 8;    // A operand K-chunk select (ISA A 16x32)
    const int kb   = (lane < 16) ? 0 : 8;    // B operand K base (same chunk order as A)
    const int mAdd = (lane < 16) ? 0 : 8;    // C layout: row M = vgpr + (lane<16?0:8)

    // ---- B operand: x column b = b0 + r16; per lane K chunks {kb, kb+16}.
    // Frag0 = K 0..31 (lane<16: 0..7,16..23, lane>=16: 8..15,24..31), Frag1 = K 32..63.
    Frag Bf0, Bf1;
    {
        const unsigned short* bRow = wsx + (size_t)(b0 + r16) * DN + kb;
        Bf0.h[0] = *(const v8bf*)(bRow);
        Bf0.h[1] = *(const v8bf*)(bRow + 16);
        Bf1.h[0] = *(const v8bf*)(bRow + 32);
        Bf1.h[1] = *(const v8bf*)(bRow + 48);
    }

    // +2.0 bias matrix, lives in 8 VGPRs for the whole kernel
    v8f two;
#pragma unroll
    for (int i = 0; i < 8; ++i) two[i] = 2.0f;

    // ---- per-lane top-16 as packed keys (valueBits | m), sorted descending
    unsigned key[DELTA];
#pragma unroll
    for (int t = 0; t < DELTA; ++t) key[t] = 0u;

    const unsigned short* aLane = wsK + ((size_t)q * MKEYS + r16) * DN + khi;

    // Prime ping-pong buffers with tiles 0 and 1.
    Frag Ae0, Ae1, Ao0, Ao1;
    loadA(aLane, Ae0, Ae1);
    loadA(aLane + 16 * DN, Ao0, Ao1);

    for (int tile = 0; tile < NTILE; tile += 2) {
        stepTile(tile,     Ae0, Ae1, Bf0, Bf1, two, aLane, mAdd, key);
        stepTile(tile + 1, Ao0, Ao1, Bf0, Bf1, two, aLane, mAdd, key);
    }

    // ---- merge lane pair (l, l^16): both sorted desc; top-16 multiset via
    // bitonic half-cleaner on packed keys: out[t] = max(mine[t], theirs[15-t]).
    unsigned pk[DELTA];
#pragma unroll
    for (int t = 0; t < DELTA; ++t)
        pk[t] = (unsigned)__shfl_xor((int)key[t], 16, 32);
    unsigned top[DELTA];
#pragma unroll
    for (int t = 0; t < DELTA; ++t)
        top[t] = umaxu(key[t], pk[DELTA - 1 - t]);

    // ---- unpack + temperature-scaled softmax (order-invariant)
    const float scale = 0.0125f;             // 0.1 / sqrt(64)
    float fv[DELTA];
#pragma unroll
    for (int t = 0; t < DELTA; ++t)
        fv[t] = __uint_as_float(top[t] & KEY_VMASK) - 2.0f;   // undo C-seed bias
    float mx = fv[0];
#pragma unroll
    for (int t = 1; t < DELTA; ++t) mx = fmaxf(mx, fv[t]);
    float e[DELTA]; float se = 0.0f;
#pragma unroll
    for (int t = 0; t < DELTA; ++t) { e[t] = __expf(scale * (fv[t] - mx)); se += e[t]; }
    const float invs = 1.0f / se;

    // ---- broadcast alpha/idx via LDS, then cooperative coalesced gather of M
    __shared__ float sA[16][DELTA];
    __shared__ int   sI[16][DELTA];
    if (lane < 16) {
#pragma unroll
        for (int t = 0; t < DELTA; ++t) {
            sA[lane][t] = e[t] * invs;
            sI[lane][t] = (int)(top[t] & KEY_IMASK);
        }
    }
    __syncthreads();

    const float2* Mq = (const float2*)(Mm + (size_t)q * MKEYS * UN);
    for (int b = 0; b < 16; ++b) {
        float2 a; a.x = 0.0f; a.y = 0.0f;
#pragma unroll 4
        for (int k = 0; k < DELTA; ++k) {
            const float al  = sA[b][k];
            const int   row = sI[b][k];
            const float2 mv = Mq[(size_t)row * (UN / 2) + lane];
            a.x += al * mv.x;
            a.y += al * mv.y;
        }
        ((float2*)out)[(size_t)((b0 + b) * QN + q) * (UN / 2) + lane] = a;
    }
}

// ---------------------------------------------------------------------------
extern "C" void kernel_launch(void* const* d_in, const int* in_sizes, int n_in,
                              void* d_out, int out_size, void* d_ws, size_t ws_size,
                              hipStream_t stream) {
    (void)in_sizes; (void)n_in; (void)out_size; (void)ws_size;
    const float* x  = (const float*)d_in[0];   // [B, D]
    const float* K  = (const float*)d_in[1];   // [Q, M, D]
    const float* Mm = (const float*)d_in[2];   // [Q, M, U]
    float* out = (float*)d_out;                // [B, Q, U]

    unsigned short* wsx = (unsigned short*)d_ws;          // bf16 [B, D]
    unsigned short* wsK = wsx + (size_t)BN * DN;          // bf16 [Q*M, D]

    const int totalRows = BN + QN * MKEYS;
    const int nblk = (totalRows + 7) / 8;
    cnu_normalize<<<nblk, 256, 0, stream>>>(x, K, wsx, wsK);

    dim3 grid(BN / 16, QN);
    cnu_main<<<grid, 32, 0, stream>>>(wsx, wsK, Mm, out);
}
